// VAE_Hist_GLM4_85615878078928
// MI455X (gfx1250) — matrix operand order repaired and checked by reading, not compiled, not run on hardware
//
#include <hip/hip_runtime.h>
#include <math.h>

typedef __attribute__((ext_vector_type(16))) _Float16 v16h;
typedef __attribute__((ext_vector_type(8)))  float    v8f;

#define SUB 20
#define KLEN 100
#define TILE 256

// ---------------------------------------------------------------------------
// Kernel 1: build alpha kernels (e/i/spk), emit out_filters [40 x 100].
// ---------------------------------------------------------------------------
__global__ void build_kernels(const float* __restrict__ W_syn,
                              const float* __restrict__ Tau_syn,
                              const float* __restrict__ Delta_syn,
                              const float* __restrict__ W_spk,
                              const float* __restrict__ Tau_spk,
                              float* __restrict__ out_filters,  // d_out + 2*T
                              float* __restrict__ ek_ws,        // [20*100]
                              float* __restrict__ spk_ws) {     // [100]
  const int tid = threadIdx.x;
  for (int i = tid; i < SUB * KLEN; i += blockDim.x) {
    const int s = i / KLEN, k = i % KLEN;
    const float te  = fmaxf((float)k - Delta_syn[s * 2 + 0], 0.f);
    const float tau_e = Tau_syn[s * 2 + 0];
    const float tte = te / (tau_e * tau_e);
    const float ev  = tte * expf(-tte) * W_syn[s * 2 + 0];
    const float ti  = fmaxf((float)k - Delta_syn[s * 2 + 1], 0.f);
    const float tau_i = Tau_syn[s * 2 + 1];
    const float tti = ti / (tau_i * tau_i);
    const float iv  = tti * expf(-tti) * W_syn[s * 2 + 1];
    out_filters[i] = ev;                    // rows 0..19 = e_kern
    out_filters[SUB * KLEN + i] = iv;       // rows 20..39 = i_kern
    ek_ws[i] = ev;
  }
  for (int k = tid; k < KLEN; k += blockDim.x) {
    const float ts = Tau_spk[0];
    const float tt = (float)k / (ts * ts);
    spk_ws[k] = tt * expf(-tt) * W_spk[0] * W_spk[0];
  }
}

// ---------------------------------------------------------------------------
// Kernel 2: syn_e = S_e @ C_syn_e^T via v_wmma_f32_16x16x32_f16.
// Spikes are 0/1 with fan-in 25 per subunit, so f16 inputs/products are exact.
// 8 waves/block; each wave owns a 16-row (t) x 32-col (subunit) D tile.
// B = C_syn_e^T staged once per block in LDS as f16, zero-padded to [32][512].
//
// Guard-free streaming:
//  - row OOB   : clamp row pointer; a garbage A-row only feeds its own D-row,
//                which is never stored.
//  - col OOB   : B rows k>=NE are zero in LDS, so A garbage there contributes 0;
//                loads just need in-bounds addresses (clamped in tail chunk).
// ---------------------------------------------------------------------------
__global__ void syn_gemm_wmma(const float* __restrict__ S_e,
                              const float* __restrict__ C_syn_e,
                              float* __restrict__ syn,  // [T][20]
                              int T, int NE) {
  __shared__ __align__(32) _Float16 Bt[32][512];  // Bt[n][k] = C_syn_e[n*NE+k]
  const int tid = threadIdx.x;
  for (int i = tid; i < 32 * 512; i += blockDim.x) {
    const int n = i >> 9, k = i & 511;
    const float v = (n < SUB && k < NE) ? C_syn_e[n * NE + k] : 0.f;
    Bt[n][k] = (_Float16)v;
  }
  __syncthreads();

  const int wave = tid >> 5;
  const int lane = tid & 31;
  const int l15  = lane & 15;
  const bool hi  = lane >= 16;
  const long t0   = (long)blockIdx.x * 128 + (long)wave * 16;
  long trow = t0 + l15;                     // A: both half-waves carry M=0..15
  if (trow > (long)T - 1) trow = (long)T - 1;  // clamp: OOB rows never stored
  const float* rowp = S_e + (size_t)trow * (size_t)NE;

  // A fragment K-offsets (contiguous, matching B):
  //   lanes 0-15 : halves 0..7 -> K=base+0..7,   halves 8..15 -> K=base+8..15
  //   lanes 16-31: halves 0..7 -> K=base+16..23, halves 8..15 -> K=base+24..31
  const int o0 = hi ? 16 : 0;
  const int o1 = hi ? 24 : 8;

  v8f acc0 = {};
  v8f acc1 = {};
  const int kchunks = (NE + 31) >> 5;       // 16 for NE=500
  const int kfull   = NE >> 5;              // 15 fully in-bounds chunks

  if ((NE & 3) == 0) {
    // ---- fast path: unconditional 16B-aligned float4 loads -----------------
#pragma unroll 3
    for (int kc = 0; kc < kfull; ++kc) {
      const int base = kc << 5;
      const float4 pa0 = *(const float4*)(rowp + base + o0);
      const float4 pa1 = *(const float4*)(rowp + base + o0 + 4);
      const float4 pb0 = *(const float4*)(rowp + base + o1);
      const float4 pb1 = *(const float4*)(rowp + base + o1 + 4);
      v16h a;
      a[0]  = (_Float16)pa0.x; a[1]  = (_Float16)pa0.y;
      a[2]  = (_Float16)pa0.z; a[3]  = (_Float16)pa0.w;
      a[4]  = (_Float16)pa1.x; a[5]  = (_Float16)pa1.y;
      a[6]  = (_Float16)pa1.z; a[7]  = (_Float16)pa1.w;
      a[8]  = (_Float16)pb0.x; a[9]  = (_Float16)pb0.y;
      a[10] = (_Float16)pb0.z; a[11] = (_Float16)pb0.w;
      a[12] = (_Float16)pb1.x; a[13] = (_Float16)pb1.y;
      a[14] = (_Float16)pb1.z; a[15] = (_Float16)pb1.w;
      const int kb = base + (hi ? 16 : 0);
      const v16h b0 = *(const v16h*)(&Bt[l15][kb]);
      const v16h b1 = *(const v16h*)(&Bt[16 + l15][kb]);
      acc0 = __builtin_amdgcn_wmma_f32_16x16x32_f16(false, a, false, b0,
                                                    (short)0, acc0, false, false);
      acc1 = __builtin_amdgcn_wmma_f32_16x16x32_f16(false, a, false, b1,
                                                    (short)0, acc1, false, false);
    }
    // ---- tail chunk: clamped, branch-free scalar loads ---------------------
    for (int kc = kfull; kc < kchunks; ++kc) {
      const int base = kc << 5;
      v16h a;
#pragma unroll
      for (int i = 0; i < 8; ++i) {
        int k0 = base + o0 + i; if (k0 > NE - 1) k0 = NE - 1;
        int k1 = base + o1 + i; if (k1 > NE - 1) k1 = NE - 1;
        a[i]     = (_Float16)rowp[k0];   // garbage beyond NE hits B==0
        a[i + 8] = (_Float16)rowp[k1];
      }
      const int kb = base + (hi ? 16 : 0);
      const v16h b0 = *(const v16h*)(&Bt[l15][kb]);
      const v16h b1 = *(const v16h*)(&Bt[16 + l15][kb]);
      acc0 = __builtin_amdgcn_wmma_f32_16x16x32_f16(false, a, false, b0,
                                                    (short)0, acc0, false, false);
      acc1 = __builtin_amdgcn_wmma_f32_16x16x32_f16(false, a, false, b1,
                                                    (short)0, acc1, false, false);
    }
  } else {
    // ---- generic path (NE not multiple of 4): clamped scalar loads ---------
    for (int kc = 0; kc < kchunks; ++kc) {
      const int base = kc << 5;
      v16h a;
#pragma unroll
      for (int i = 0; i < 8; ++i) {
        int k0 = base + o0 + i; if (k0 > NE - 1) k0 = NE - 1;
        int k1 = base + o1 + i; if (k1 > NE - 1) k1 = NE - 1;
        a[i]     = (_Float16)rowp[k0];
        a[i + 8] = (_Float16)rowp[k1];
      }
      const int kb = base + (hi ? 16 : 0);
      const v16h b0 = *(const v16h*)(&Bt[l15][kb]);
      const v16h b1 = *(const v16h*)(&Bt[16 + l15][kb]);
      acc0 = __builtin_amdgcn_wmma_f32_16x16x32_f16(false, a, false, b0,
                                                    (short)0, acc0, false, false);
      acc1 = __builtin_amdgcn_wmma_f32_16x16x32_f16(false, a, false, b1,
                                                    (short)0, acc1, false, false);
    }
  }

  // D layout: lane in [0,16): N=lane, M=r; lane in [16,32): N=lane-16, M=r+8
#pragma unroll
  for (int r = 0; r < 8; ++r) {
    const long m = t0 + r + (hi ? 8 : 0);
    if (m < (long)T) {
      if (l15 < SUB)      syn[m * SUB + l15]      = acc0[r];
      if (16 + l15 < SUB) syn[m * SUB + 16 + l15] = acc1[r];
    }
  }
}

// ---------------------------------------------------------------------------
// Kernel 3: causal conv (syn = 2*filt_e), tree recursion, sigmoid, spike conv.
// ---------------------------------------------------------------------------
__global__ void conv_tree_out(const float* __restrict__ syn,   // [T][20]
                              const float* __restrict__ ek,    // [20][100]
                              const float* __restrict__ spk,   // [100]
                              const float* __restrict__ Z,     // [T]
                              const float* __restrict__ C_den, // [20][20]
                              const float* __restrict__ W_sub,
                              const float* __restrict__ W_plex,
                              const float* __restrict__ Theta,
                              float* __restrict__ out, int T) {
  __shared__ float sSyn[(TILE + KLEN - 1) * SUB];  // t0-99 .. t0+TILE-1
  __shared__ float sZ[TILE + KLEN];                // t0-100 .. t0+TILE-1
  __shared__ float sEk[SUB * KLEN];
  __shared__ float sSpk[KLEN];
  __shared__ float sCden[SUB * SUB];
  __shared__ float sWsub2[SUB], sWplex2[SUB], sTheta[SUB];

  const int tid = threadIdx.x;
  const long t0 = (long)blockIdx.x * TILE;
  for (int i = tid; i < (TILE + KLEN - 1) * SUB; i += blockDim.x) {
    const int row = i / SUB, s = i % SUB;
    const long t = t0 - (KLEN - 1) + row;
    sSyn[i] = (t >= 0 && t < (long)T) ? syn[t * SUB + s] : 0.f;
  }
  for (int i = tid; i < TILE + KLEN; i += blockDim.x) {
    const long t = t0 - KLEN + i;
    sZ[i] = (t >= 0 && t < (long)T) ? Z[t] : 0.f;
  }
  for (int i = tid; i < SUB * KLEN; i += blockDim.x) sEk[i] = ek[i];
  for (int i = tid; i < KLEN; i += blockDim.x) sSpk[i] = spk[i];
  for (int i = tid; i < SUB * SUB; i += blockDim.x) sCden[i] = C_den[i];
  for (int i = tid; i < SUB; i += blockDim.x) {
    const float ws = W_sub[i];  sWsub2[i]  = ws * ws;
    const float wp = W_plex[i]; sWplex2[i] = wp * wp;
    sTheta[i] = Theta[i];
  }
  __syncthreads();

  const long tt = t0 + tid;
  if (tt >= (long)T) return;

  // filt_e[s] = sum_j e_kern[s,j] * syn_e[tt-j, s]
  float f[SUB];
#pragma unroll
  for (int s = 0; s < SUB; ++s) f[s] = 0.f;
  for (int j = 0; j < KLEN; ++j) {
    const float* srow = &sSyn[(tid + KLEN - 1 - j) * SUB];
#pragma unroll
    for (int s = 0; s < SUB; ++s) f[s] = fmaf(sEk[s * KLEN + j], srow[s], f[s]);
  }
  // tree: leaves -> root (children have higher index); syn = 2*filt_e
  float so[SUB];
#pragma unroll
  for (int s = 0; s < SUB; ++s) so[s] = 0.f;
#pragma unroll
  for (int s = SUB - 1; s >= 0; --s) {
    float anc = 0.f;
#pragma unroll
    for (int c = 0; c < SUB; ++c)
      anc = fmaf(sCden[s * SUB + c] * sWsub2[c], so[c], anc);
    const float sub_in = 2.f * f[s] + sTheta[s] + anc;
    so[s] = tanhf(sub_in) * sWplex2[s];
  }
  out[(size_t)T + tt] = 1.f / (1.f + expf(-so[0]));  // final_Z

  // final_V[tt] = sum_j spk_kern[j] * Z[tt-1-j]   (t=0 -> 0 via zero padding)
  float v = 0.f;
  for (int j = 0; j < KLEN; ++j)
    v = fmaf(sSpk[j], sZ[tid + KLEN - 1 - j], v);
  out[tt] = v;                                       // final_V
}

// ---------------------------------------------------------------------------
extern "C" void kernel_launch(void* const* d_in, const int* in_sizes, int n_in,
                              void* d_out, int out_size, void* d_ws, size_t ws_size,
                              hipStream_t stream) {
  const float* S_e       = (const float*)d_in[0];
  // d_in[1] = S_i : dead in the reference (filt_i bug) -> never read
  const float* Z         = (const float*)d_in[2];
  const float* C_den     = (const float*)d_in[3];
  const float* C_syn_e   = (const float*)d_in[4];
  const float* W_syn     = (const float*)d_in[6];
  const float* Tau_syn   = (const float*)d_in[7];
  const float* Delta_syn = (const float*)d_in[8];
  const float* W_plex    = (const float*)d_in[9];
  const float* W_sub     = (const float*)d_in[10];
  const float* W_spk     = (const float*)d_in[11];
  const float* Tau_spk   = (const float*)d_in[12];
  const float* Theta     = (const float*)d_in[13];

  const int T  = in_sizes[2];             // 100000
  const int NE = in_sizes[0] / T;         // 500

  float* out = (float*)d_out;             // [final_V(T) | final_Z(T) | filters(4000)]
  float* syn = (float*)d_ws;              // [T][20] floats
  float* ek  = syn + (size_t)T * SUB;     // [20*100]
  float* spk = ek + SUB * KLEN;           // [100]

  build_kernels<<<1, 256, 0, stream>>>(W_syn, Tau_syn, Delta_syn, W_spk,
                                       Tau_spk, out + 2 * (size_t)T, ek, spk);

  const int gb = (T + 127) / 128;         // 128 t-rows per block (8 waves x 16)
  syn_gemm_wmma<<<gb, 256, 0, stream>>>(S_e, C_syn_e, syn, T, NE);

  const int gc = (T + TILE - 1) / TILE;
  conv_tree_out<<<gc, TILE, 0, stream>>>(syn, ek, spk, Z, C_den, W_sub,
                                         W_plex, Theta, out, T);
}
